// FP8StaticLinear_1202590843022
// MI455X (gfx1250) — 8-bit matrix operand order repaired and checked by reading, compiled, not run on hardware
//
#include <hip/hip_runtime.h>
#include <stdint.h>
#include <stddef.h>


typedef __attribute__((ext_vector_type(16))) int   v16i;
typedef __attribute__((ext_vector_type(8)))  float v8f;
typedef __attribute__((ext_vector_type(4)))  int   v4i;
typedef __attribute__((ext_vector_type(2)))  int   v2i;
typedef __attribute__((ext_vector_type(4)))  float v4f;

typedef __attribute__((address_space(1))) v4i g_v4i;
typedef __attribute__((address_space(3))) v4i l_v4i;

#define BM 128
#define BN 128
#define BK 128
#define LDSS 144
#define TILE_BYTES (BM * LDSS)
#define SMEM_BYTES (4 * TILE_BYTES)

#if defined(__has_builtin)
#if __has_builtin(__builtin_amdgcn_cvt_pk_fp8_f32)
#define HAVE_CVT_PK_FP8 1
#endif
#if __has_builtin(__builtin_amdgcn_global_load_async_to_lds_b128)
#define HAVE_ASYNC_LDS 1
#endif
#if __has_builtin(__builtin_amdgcn_s_wait_asynccnt)
#define HAVE_WAIT_ASYNC 1
#endif
#endif

#define F8MAX 448.0f

__device__ __forceinline__ float clamp_f8(float x) {
    return fminf(fmaxf(x, -F8MAX), F8MAX);
}

#ifndef HAVE_CVT_PK_FP8
__device__ __forceinline__ unsigned f32_to_e4m3_sw(float x) {
    unsigned sgn = (__float_as_uint(x) >> 31) << 7;
    float ax = fabsf(x);
    if (!(ax > 0.0f)) return sgn;
    if (ax >= F8MAX) return sgn | 0x7E;
    int e;
    float m = frexpf(ax, &e);
    int E = e - 1 + 7;
    unsigned bits;
    if (E >= 1) {
        unsigned q = (unsigned)(m * 16.0f + 0.5f);
        if (q >= 16u) { q >>= 1; E++; }
        if (E > 15 || (E == 15 && (q & 7u) == 7u)) { E = 15; q = 14u; }
        bits = ((unsigned)E << 3) | (q & 7u);
    } else {
        unsigned q = (unsigned)(ax * 512.0f + 0.5f);
        bits = (q > 7u) ? 8u : q;
    }
    return sgn | bits;
}
#endif

__device__ __forceinline__ unsigned pack4_e4m3(float a, float b, float c, float d) {
#ifdef HAVE_CVT_PK_FP8
    int v = 0;
    v = __builtin_amdgcn_cvt_pk_fp8_f32(clamp_f8(a), clamp_f8(b), v, false);
    v = __builtin_amdgcn_cvt_pk_fp8_f32(clamp_f8(c), clamp_f8(d), v, true);
    return (unsigned)v;
#else
    return  f32_to_e4m3_sw(a)
         | (f32_to_e4m3_sw(b) << 8)
         | (f32_to_e4m3_sw(c) << 16)
         | (f32_to_e4m3_sw(d) << 24);
#endif
}

__device__ __forceinline__ void async_copy_b128(const uint8_t* g, uint8_t* l) {
#ifdef HAVE_ASYNC_LDS
    __builtin_amdgcn_global_load_async_to_lds_b128(
        (g_v4i*)(uintptr_t)g,
        (l_v4i*)(uintptr_t)l,
        0, 0);
#else
    *(v4i*)l = *(const v4i*)g;
#endif
}

__device__ __forceinline__ void wait_async_lds() {
#ifdef HAVE_ASYNC_LDS
#ifdef HAVE_WAIT_ASYNC
    __builtin_amdgcn_s_wait_asynccnt(0);
#else
    asm volatile("s_wait_asynccnt 0" ::: "memory");
#endif
#endif
}

__device__ __forceinline__ void load_a_regs(const float* __restrict__ X, int K,
                                            int blockM0, int k0, int wave, int lane,
                                            v4f a[16]) {
#pragma unroll
    for (int i = 0; i < 16; ++i) {
        const int row = i * 8 + wave;
        a[i] = *(const v4f*)(X + (size_t)(blockM0 + row) * K + k0 + lane * 4);
    }
}

__device__ __forceinline__ void quant_store_a(const v4f a[16], uint8_t* __restrict__ As,
                                              int wave, int lane, float rcp) {
#pragma unroll
    for (int i = 0; i < 16; ++i) {
        const int row = i * 8 + wave;
        const unsigned p = pack4_e4m3(a[i].x * rcp, a[i].y * rcp,
                                      a[i].z * rcp, a[i].w * rcp);
        *(unsigned*)(As + row * LDSS + lane * 4) = p;
    }
}

__device__ __forceinline__ void copy_b_tile(const uint8_t* __restrict__ W, int K,
                                            int blockN0, int k0, int tid,
                                            uint8_t* __restrict__ Bs) {
#pragma unroll
    for (int i = 0; i < 4; ++i) {
        const int idx = i * 256 + tid;
        const int row = idx >> 3;
        const int col = (idx & 7) * 16;
        async_copy_b128(W + (size_t)(blockN0 + row) * K + k0 + col,
                        Bs + row * LDSS + col);
    }
}

__global__ __launch_bounds__(256)
void fp8_static_linear_kernel(const float*   __restrict__ X,
                              const uint8_t* __restrict__ W,
                              const float*   __restrict__ wscale,
                              const float*   __restrict__ iscale,
                              const float*   __restrict__ bias,
                              float*         __restrict__ out,
                              int M, int N, int K)
{
    extern __shared__ uint8_t smem[];

    const int tid   = (int)threadIdx.x;
    const int lane  = tid & 31;
    const int wave  = tid >> 5;
    const int waveM = wave >> 1;
    const int waveN = wave & 1;
    const int lr    = lane & 15;
    const int lh    = lane >> 4;

    const int blockN0 = (int)blockIdx.x * BN;
    const int blockM0 = (int)blockIdx.y * BM;

    const float in_s   = iscale[0];
    const float w_s    = wscale[0];
    const float rcp_in = 1.0f / in_s;

    v8f acc[2][4];
#pragma unroll
    for (int ms = 0; ms < 2; ++ms)
#pragma unroll
        for (int ns = 0; ns < 4; ++ns)
            acc[ms][ns] = (v8f){0.f, 0.f, 0.f, 0.f, 0.f, 0.f, 0.f, 0.f};

    v4f areg[16];
    load_a_regs(X, K, blockM0, 0, wave, lane, areg);
    copy_b_tile(W, K, blockN0, 0, tid, smem + 2 * TILE_BYTES);
    quant_store_a(areg, smem, wave, lane, rcp_in);
    wait_async_lds();
    __syncthreads();

    int cur = 0;
    for (int k0 = 0; k0 < K; k0 += BK) {
        const int nxt = cur ^ 1;
        const bool has_next = (k0 + BK) < K;

        if (has_next) {
            load_a_regs(X, K, blockM0, k0 + BK, wave, lane, areg);
            copy_b_tile(W, K, blockN0, k0 + BK, tid,
                        smem + (2 + nxt) * TILE_BYTES);
        }

        const uint8_t* As = smem + cur * TILE_BYTES;
        const uint8_t* Bs = smem + (2 + cur) * TILE_BYTES;

        v16i afrag[2];
#pragma unroll
        for (int ms = 0; ms < 2; ++ms) {
            const int row = waveM * 32 + ms * 16 + lr;
            const uint8_t* p = &As[row * LDSS + lh * 16];
#pragma unroll
            for (int q = 0; q < 8; ++q) {
                const v2i t = *(const v2i*)(p + (q >> 1) * 32 + (q & 1) * 8);
                afrag[ms][2 * q]     = t.x;
                afrag[ms][2 * q + 1] = t.y;
            }
        }
#pragma unroll
        for (int ns = 0; ns < 4; ++ns) {
            const int col = waveN * 64 + ns * 16 + lr;
            const uint8_t* p = &Bs[col * LDSS + lh * 16];
            v16i bfrag;
#pragma unroll
            for (int g = 0; g < 4; ++g) {
                const v4i t = *(const v4i*)(p + g * 32);
                bfrag[4 * g]     = t.x;
                bfrag[4 * g + 1] = t.y;
                bfrag[4 * g + 2] = t.z;
                bfrag[4 * g + 3] = t.w;
            }
#pragma unroll
            for (int ms = 0; ms < 2; ++ms) {
                acc[ms][ns] = __builtin_amdgcn_wmma_f32_16x16x128_fp8_fp8(
                    afrag[ms], bfrag, (short)0, acc[ms][ns],
                     false,  false);
            }
        }

        if (has_next) {
            quant_store_a(areg, smem + nxt * TILE_BYTES, wave, lane, rcp_in);
            wait_async_lds();
        }
        __syncthreads();
        cur = nxt;
    }

    const float scale = in_s * w_s;
#pragma unroll
    for (int ms = 0; ms < 2; ++ms) {
#pragma unroll
        for (int ns = 0; ns < 4; ++ns) {
            const int gm0 = blockM0 + waveM * 32 + ms * 16 + lh * 8;
            const int gn  = blockN0 + waveN * 64 + ns * 16 + lr;
            const float bv = bias[gn];
#pragma unroll
            for (int r = 0; r < 8; ++r) {
                __builtin_nontemporal_store(acc[ms][ns][r] * scale + bv,
                                            out + (size_t)(gm0 + r) * N + gn);
            }
        }
    }
}

extern "C" void kernel_launch(void* const* d_in, const int* in_sizes, int n_in,
                              void* d_out, int out_size, void* d_ws, size_t ws_size,
                              hipStream_t stream) {
    (void)n_in; (void)out_size; (void)d_ws; (void)ws_size;
    const float*   x    = (const float*)d_in[0];
    const uint8_t* qw   = (const uint8_t*)d_in[1];
    const float*   wsc  = (const float*)d_in[2];
    const float*   isc  = (const float*)d_in[3];
    const float*   bias = (const float*)d_in[4];
    float*         out  = (float*)d_out;

    const int OUT = in_sizes[4];
    const int IN  = in_sizes[1] / OUT;
    const int M   = in_sizes[0] / IN;

    dim3 grid((unsigned)(OUT / BN), (unsigned)(M / BM));
    fp8_static_linear_kernel<<<grid, 256, SMEM_BYTES, stream>>>(x, qw, wsc, isc, bias,
                                                                out, M, OUT, IN);
}
